// SelectiveMambaBlock_11931419148454
// MI455X (gfx1250) — matrix operand order repaired and checked by reading, not compiled, not run on hardware
//
#include <hip/hip_runtime.h>
#include <hip/hip_bf16.h>

// ---------------------------------------------------------------------------
// Mamba selective-scan block for MI455X (gfx1250, wave32, WMMA + TDM).
// Big GEMMs in bf16 WMMA (f32 accum) -> memory-bound at 23.3 TB/s.
// Activation panels staged into LDS by the Tensor Data Mover (TENSORcnt);
// weights pre-transposed at conversion so B operands are K-contiguous loads.
// Scan / norm / conv / softplus in f32 VALU.
// ---------------------------------------------------------------------------

typedef __attribute__((ext_vector_type(16))) __bf16 v16bf;
typedef __attribute__((ext_vector_type(8)))  __bf16 v8bf;
typedef __attribute__((ext_vector_type(8)))  float  v8f;

#if __has_builtin(__builtin_amdgcn_tensor_load_to_lds)
#define HAVE_TDM 1
typedef unsigned int v4u __attribute__((ext_vector_type(4)));
typedef int          v4i __attribute__((ext_vector_type(4)));
typedef int          v8i __attribute__((ext_vector_type(8)));
#else
#define HAVE_TDM 0
#endif

#define D_MODEL 1024
#define D_STATE 16
#define D_CONV  4
#define D_INNER 2048
#define DT_RANK 64
#define B_SZ    2
#define L_SEQ   4096
#define NTOK    (B_SZ * L_SEQ)   // 8192 tokens
#define KCHUNK  128              // A-panel depth staged per TDM step

// ---------------------------------------------------------------------------
// f32 -> bf16 weight conversion WITH transpose: dst[n*K + k] = src[k*N + n]
// (makes the WMMA B operand two K-contiguous 16B runs per lane)
// ---------------------------------------------------------------------------
__global__ void cvt_bf16_T_kernel(const float* __restrict__ src,
                                  __bf16* __restrict__ dst, int K, int N) {
    const size_t i = (size_t)blockIdx.x * 256 + threadIdx.x;   // over K*N
    const int n = (int)(i / K);
    const int k = (int)(i % K);
    dst[i] = (__bf16)src[(size_t)k * N + n];
}

// ---------------------------------------------------------------------------
// LayerNorm over D_MODEL, one block (256 thr) per token; bf16 output for WMMA A
// ---------------------------------------------------------------------------
__global__ void layernorm_kernel(const float* __restrict__ x,
                                 const float* __restrict__ w,
                                 const float* __restrict__ b,
                                 __bf16* __restrict__ xn) {
    __shared__ float row[D_MODEL];
    __shared__ float red[2][8];
    const int tok = blockIdx.x, tid = threadIdx.x;
    float s = 0.f, s2 = 0.f;
    for (int i = tid; i < D_MODEL; i += 256) {
        float v = x[(size_t)tok * D_MODEL + i];
        row[i] = v; s += v; s2 += v * v;
    }
    for (int m = 1; m < 32; m <<= 1) {
        s  += __shfl_xor(s,  m, 32);
        s2 += __shfl_xor(s2, m, 32);
    }
    if ((tid & 31) == 0) { red[0][tid >> 5] = s; red[1][tid >> 5] = s2; }
    __syncthreads();
    if (tid == 0) {
        float a = 0.f, c = 0.f;
        for (int i = 0; i < 8; ++i) { a += red[0][i]; c += red[1][i]; }
        red[0][0] = a; red[1][0] = c;
    }
    __syncthreads();
    const float mu  = red[0][0] * (1.f / D_MODEL);
    const float var = red[1][0] * (1.f / D_MODEL) - mu * mu;
    const float inv = rsqrtf(var + 1e-5f);
    for (int i = tid; i < D_MODEL; i += 256)
        xn[(size_t)tok * D_MODEL + i] = (__bf16)((row[i] - mu) * inv * w[i] + b[i]);
}

// ---------------------------------------------------------------------------
// bf16 WMMA GEMM: D[M,N] = A[M,K] * BT^T, f32 accumulate.
//   A  : row-major [M,K] bf16 (activations)
//   BT : TRANSPOSED weights [N,K] bf16 -> B operand is K-contiguous per lane
// Block = 8 waves computes a 64(M) x 128(N) panel; waves share one 64xKCHUNK
// A panel staged in LDS by ONE tensor_load_to_lds per chunk (wave 0), guarded
// by s_wait_tensorcnt + block barrier.  Per wave: 4 M-subtiles x 16 cols.
// A operand (16x32 bf16 layout): lane(half h,row m) holds K={k0+8h..+7} and
// {k0+16+8h..+7} -> two ds_load_b128 from the LDS panel.
// B operand: lane(h,col) holds K={k0+8h..+7} and {k0+16+8h..+7} -> two 16B runs from BT.
// mode 0: split store (col<D_INNER -> out0, else out1), both [M,D_INNER]
// mode 1: out0[row*N+col] = resid[row*N+col] + acc
// ---------------------------------------------------------------------------
__global__ void gemm_bf16_kernel(const __bf16* __restrict__ A,
                                 const __bf16* __restrict__ BT,
                                 int K, int N, int nGroups, int mode,
                                 const float* __restrict__ resid,
                                 float* __restrict__ out0,
                                 float* __restrict__ out1) {
    __shared__ __align__(16) __bf16 ldsA[64][KCHUNK];      // 16 KB / block
    const int w    = threadIdx.x >> 5;
    const int lane = threadIdx.x & 31;
    const int tid  = threadIdx.x;
    const int mg = blockIdx.x / nGroups;        // 64-row M group
    const int ng = blockIdx.x % nGroups;        // 128-col N group
    const int tn = ng * 8 + w;                  // this wave's 16-col tile
    const int half = lane >> 4, lr = lane & 15;
    const int col = tn * 16 + lr;

    v8f acc[4] = {};
    const __bf16* brow = BT + (size_t)col * K;  // K-contiguous column data
    const __bf16* apan = A + (size_t)(mg * 64) * K;

    for (int kc = 0; kc < K; kc += KCHUNK) {
        __syncthreads();   // previous chunk's LDS reads drained before overwrite
#if HAVE_TDM
        if (w == 0) {
            // ---- Tensor DMA descriptor (D#), ISA ch.8: 2D tensor,
            //      tile = KCHUNK(x, contiguous K) x 64(y, rows)
            unsigned long long ga = (unsigned long long)(const void*)&apan[kc];
            unsigned int lds =
                (unsigned int)(unsigned long long)(const void*)&ldsA[0][0];
            v4u g0;
            g0[0] = 1u;                                   // count=1 (user D#)
            g0[1] = lds;                                  // lds_addr
            g0[2] = (unsigned int)(ga & 0xFFFFFFFFu);     // global_addr[31:0]
            g0[3] = (unsigned int)((ga >> 32) & 0x01FFFFFFu) | (2u << 30); // type=2
            v8i g1;
            const unsigned mrem = (unsigned)(NTOK - mg * 64);
            g1[0] = 0x00010000;                           // data_size=1 (2 bytes)
            g1[1] = (int)(((unsigned)K & 0xFFFFu) << 16); // tensor_dim0[15:0]=K
            g1[2] = (int)((mrem & 0xFFFFu) << 16)         // tensor_dim1=rows left
                  | (int)((unsigned)K >> 16);             //  | tensor_dim0 hi
            g1[3] = (int)((unsigned)KCHUNK << 16);        // tile_dim0=128 (K)
            g1[4] = 64;                                   // tile_dim1=64 (rows)
            g1[5] = (int)K;                               // tensor_dim0_stride=K
            g1[6] = 0;
            g1[7] = 0;
            v4i g2 = {}; v4i g3 = {};
#if defined(__clang_major__) && (__clang_major__ >= 23)
            v8i gpad = {};
            __builtin_amdgcn_tensor_load_to_lds(g0, g1, g2, g3, gpad, 0);
#else
            __builtin_amdgcn_tensor_load_to_lds(g0, g1, g2, g3, 0);
#endif
#if __has_builtin(__builtin_amdgcn_s_wait_tensorcnt)
            __builtin_amdgcn_s_wait_tensorcnt(0);
#else
            asm volatile("s_wait_tensorcnt 0x0" ::: "memory");
#endif
        }
#else
        // ---- fallback: cooperative vector copy, 64x128 bf16 = 1024 x 16B
        #pragma unroll
        for (int c = tid; c < 1024; c += 256) {
            const int r = c >> 4, seg = c & 15;
            *(v8bf*)&ldsA[r][seg * 8] =
                *(const v8bf*)&apan[(size_t)r * K + kc + seg * 8];
        }
#endif
        __syncthreads();   // panel visible to all 8 waves

        if (kc + KCHUNK < K)
            __builtin_prefetch(&brow[kc + KCHUNK], 0, 1);  // global_prefetch_b8

        // ---- 4 K-steps of 32, 4 M-subtiles each -> 16 WMMA per chunk
        #pragma unroll
        for (int ks = 0; ks < 4; ++ks) {
            // B operand: two 8-wide K runs (k0+8h, k0+16+8h) from transposed weights
            union { v16bf v; v8bf h[2]; } bu;
            bu.h[0] = *(const v8bf*)&brow[kc + ks * 32 + 8 * half];
            bu.h[1] = *(const v8bf*)&brow[kc + ks * 32 + 16 + 8 * half];
            const v16bf bv = bu.v;
            #pragma unroll
            for (int i = 0; i < 4; ++i) {
                union { v16bf v; v8bf h[2]; } av;
                av.h[0] = *(const v8bf*)&ldsA[i * 16 + lr][ks * 32 + 8 * half];
                av.h[1] = *(const v8bf*)&ldsA[i * 16 + lr][ks * 32 + 16 + 8 * half];
                acc[i] = __builtin_amdgcn_wmma_f32_16x16x32_bf16(
                             false, av.v, false, bv, (short)0, acc[i],
                             false, false);
            }
        }
    }

    // C/D layout: VGPR r -> M = r + 8*half, N = lr
    #pragma unroll
    for (int i = 0; i < 4; ++i) {
        #pragma unroll
        for (int r = 0; r < 8; ++r) {
            const int row = mg * 64 + i * 16 + r + 8 * half;
            const float v = acc[i][r];
            if (mode == 0) {
                if (col < D_INNER) out0[(size_t)row * D_INNER + col] = v;
                else               out1[(size_t)row * D_INNER + (col - D_INNER)] = v;
            } else {
                out0[(size_t)row * N + col] = resid[(size_t)row * N + col] + v;
            }
        }
    }
}

// ---------------------------------------------------------------------------
// Depthwise causal conv1d (k=4, left pad 3, cross-correlation) + bias + SiLU
// ---------------------------------------------------------------------------
__global__ void conv_silu_kernel(const float* __restrict__ xs,
                                 const float* __restrict__ cw,
                                 const float* __restrict__ cb,
                                 float* __restrict__ xc) {
    const size_t idx = (size_t)blockIdx.x * 256 + threadIdx.x;  // < NTOK*D_INNER
    const int d   = (int)(idx & (D_INNER - 1));
    const int tok = (int)(idx >> 11);
    const int l   = tok & (L_SEQ - 1);
    float acc = cb[d];
    #pragma unroll
    for (int j = 0; j < D_CONV; ++j) {
        const int li = l - (D_CONV - 1) + j;
        if (li >= 0)
            acc += cw[d * D_CONV + j] * xs[(size_t)(tok - (D_CONV - 1) + j) * D_INNER + d];
    }
    xc[idx] = acc / (1.f + __expf(-acc));   // silu
}

// ---------------------------------------------------------------------------
// x_dbl = x_conv @ W_x (N=96: [dt_low|B|C]) and delta = softplus(dt_low@W_dt + b_dt)
// One block per token; the 2048-wide row staged in LDS.
// ---------------------------------------------------------------------------
__global__ void xdbl_delta_kernel(const float* __restrict__ xc,
                                  const float* __restrict__ Wx,
                                  const float* __restrict__ Wdt,
                                  const float* __restrict__ bdt,
                                  float* __restrict__ xdbl,
                                  float* __restrict__ delta) {
    __shared__ float srow[D_INNER];
    __shared__ float sdl[DT_RANK];
    const int tok = blockIdx.x, tid = threadIdx.x;
    for (int i = tid; i < D_INNER; i += 256)
        srow[i] = xc[(size_t)tok * D_INNER + i];
    __syncthreads();
    if (tid < (DT_RANK + 2 * D_STATE)) {      // 96 outputs
        float a = 0.f;
        for (int k = 0; k < D_INNER; ++k)
            a += srow[k] * Wx[(size_t)k * (DT_RANK + 2 * D_STATE) + tid];
        xdbl[(size_t)tok * (DT_RANK + 2 * D_STATE) + tid] = a;
        if (tid < DT_RANK) sdl[tid] = a;
    }
    __syncthreads();
    for (int d = tid; d < D_INNER; d += 256) {
        float a = bdt[d];
        #pragma unroll 8
        for (int r = 0; r < DT_RANK; ++r) a += sdl[r] * Wdt[(size_t)r * D_INNER + d];
        delta[(size_t)tok * D_INNER + d] = (a > 20.f) ? a : log1pf(__expf(a));
    }
}

// ---------------------------------------------------------------------------
// Selective scan: 16 lanes per (b,d) channel, one per state n. Sequential in L.
// y reduced across the 16-lane half-wave via shfl_xor. Fused * silu(z) gate,
// bf16 output feeds the final WMMA GEMM.
// ---------------------------------------------------------------------------
__global__ void scan_kernel(const float* __restrict__ delta,
                            const float* __restrict__ u,
                            const float* __restrict__ xdbl,
                            const float* __restrict__ zb,
                            const float* __restrict__ Alog,
                            const float* __restrict__ Dv,
                            __bf16* __restrict__ yg) {
    const int gid = blockIdx.x * 256 + threadIdx.x;
    const int n  = gid & (D_STATE - 1);
    const int ch = gid >> 4;                 // 0 .. B_SZ*D_INNER-1
    const int b  = ch >> 11;
    const int d  = ch & (D_INNER - 1);
    const float An = -__expf(Alog[d * D_STATE + n]);
    const float Dp = Dv[d];
    float h = 0.f;
    const size_t tokBase = (size_t)b * L_SEQ;
    for (int l = 0; l < L_SEQ; ++l) {
        const size_t tok = tokBase + l;
        const float dt = delta[tok * D_INNER + d];
        const float ut = u[tok * D_INNER + d];
        const float Bt = xdbl[tok * 96 + DT_RANK + n];
        const float Ct = xdbl[tok * 96 + DT_RANK + D_STATE + n];
        h = __expf(dt * An) * h + (dt * Bt) * ut;
        float p = h * Ct;
        p += __shfl_xor(p, 1, 32);
        p += __shfl_xor(p, 2, 32);
        p += __shfl_xor(p, 4, 32);
        p += __shfl_xor(p, 8, 32);           // sum over 16-lane half
        if (n == 0) {
            const float zv = zb[tok * D_INNER + d];
            const float g  = zv / (1.f + __expf(-zv));
            yg[tok * D_INNER + d] = (__bf16)((p + ut * Dp) * g);
        }
    }
}

// ---------------------------------------------------------------------------
// Host-side launch
// ---------------------------------------------------------------------------
extern "C" void kernel_launch(void* const* d_in, const int* in_sizes, int n_in,
                              void* d_out, int out_size, void* d_ws, size_t ws_size,
                              hipStream_t stream) {
    (void)in_sizes; (void)n_in; (void)out_size; (void)ws_size;

    const float* x      = (const float*)d_in[0];   // [B,L,D_MODEL]
    const float* norm_w = (const float*)d_in[1];
    const float* norm_b = (const float*)d_in[2];
    const float* W_in   = (const float*)d_in[3];   // [D_MODEL, 2*D_INNER]
    const float* conv_w = (const float*)d_in[4];   // [D_INNER,1,D_CONV]
    const float* conv_b = (const float*)d_in[5];
    const float* W_x    = (const float*)d_in[6];   // [D_INNER, 96]
    const float* W_dt   = (const float*)d_in[7];   // [DT_RANK, D_INNER]
    const float* b_dt   = (const float*)d_in[8];
    const float* A_log  = (const float*)d_in[9];   // [D_INNER, D_STATE]
    const float* Dvec   = (const float*)d_in[10];
    const float* W_out  = (const float*)d_in[11];  // [D_INNER, D_MODEL]
    float* out = (float*)d_out;

    // Workspace layout (bytes)
    char* ws = (char*)d_ws;
    size_t o = 0;
    __bf16* xn_bf   = (__bf16*)(ws + o); o += (size_t)NTOK * D_MODEL * 2;       // 16 MB
    __bf16* winT_bf = (__bf16*)(ws + o); o += (size_t)D_MODEL * 2 * D_INNER * 2;// 8 MB
    __bf16* woutT_bf= (__bf16*)(ws + o); o += (size_t)D_INNER * D_MODEL * 2;    // 4 MB
    float*  xssm    = (float*)(ws + o);  o += (size_t)NTOK * D_INNER * 4;       // 64 MB
    float*  zbuf    = (float*)(ws + o);  o += (size_t)NTOK * D_INNER * 4;       // 64 MB
    float*  xconv   = (float*)(ws + o);  o += (size_t)NTOK * D_INNER * 4;       // 64 MB
    float*  xdbl    = (float*)(ws + o);  o += (size_t)NTOK * 96 * 4;            // 3 MB
    float*  delta   = (float*)(ws + o);  o += (size_t)NTOK * D_INNER * 4;       // 64 MB
    __bf16* yg_bf   = (__bf16*)(ws + o); o += (size_t)NTOK * D_INNER * 2;       // 32 MB

    // 1) weight conversions WITH transpose: [K,N] f32 -> [N,K] bf16
    cvt_bf16_T_kernel<<<(D_MODEL * 2 * D_INNER) / 256, 256, 0, stream>>>(
        W_in, winT_bf, D_MODEL, 2 * D_INNER);
    cvt_bf16_T_kernel<<<(D_INNER * D_MODEL) / 256, 256, 0, stream>>>(
        W_out, woutT_bf, D_INNER, D_MODEL);

    // 2) layernorm -> bf16 xn
    layernorm_kernel<<<NTOK, 256, 0, stream>>>(x, norm_w, norm_b, xn_bf);

    // 3) xz = xn @ W_in  (M=8192, K=1024, N=4096): 128 Mgroups * 32 Ngroups
    gemm_bf16_kernel<<<128 * 32, 256, 0, stream>>>(
        xn_bf, winT_bf, D_MODEL, 2 * D_INNER, 32, 0,
        nullptr, xssm, zbuf);

    // 4) depthwise causal conv + SiLU
    conv_silu_kernel<<<(NTOK * D_INNER) / 256, 256, 0, stream>>>(
        xssm, conv_w, conv_b, xconv);

    // 5) x_dbl + delta
    xdbl_delta_kernel<<<NTOK, 256, 0, stream>>>(xconv, W_x, W_dt, b_dt, xdbl, delta);

    // 6) selective scan + gate -> bf16 y
    scan_kernel<<<(B_SZ * D_INNER * D_STATE) / 256, 256, 0, stream>>>(
        delta, xconv, xdbl, zbuf, A_log, Dvec, yg_bf);

    // 7) out = residual + y @ W_out  (M=8192, K=2048, N=1024): 128 * 8 groups
    gemm_bf16_kernel<<<128 * 8, 256, 0, stream>>>(
        yg_bf, woutT_bf, D_INNER, D_MODEL, 8, 1,
        x, out, nullptr);
}
